// TransFrame_71270687310413
// MI455X (gfx1250) — matrix operand order repaired and checked by reading, not compiled, not run on hardware
//
#include <hip/hip_runtime.h>
#include <hip/hip_bf16.h>
#include <math.h>

// ---------------- problem constants ----------------
#define BB 2
#define HH 128
#define WW 128
#define DD 8
#define CC 64
#define NN (HH*WW)              // 16384 spatial tokens per batch
#define FF 512                  // feature dim (= D*C = heads*Dh)
#define BN (BB*NN)              // 32768 total rows
#define HEADS 4
#define DH 128                  // per-head feature dim (dim_head * D)
#define KSPAN 1024              // split-K chunk for attention GEMM
#define NCHUNK 16               // split-n chunks for column norms
static const size_t BNF = (size_t)BN * FF;   // 16,777,216 elements

// ---------------- CDNA5 WMMA types ----------------
typedef __bf16 bf16_t;
typedef bf16_t v16bf __attribute__((ext_vector_type(16)));
typedef bf16_t v8bf  __attribute__((ext_vector_type(8)));
typedef float  v8f   __attribute__((ext_vector_type(8)));

__device__ __forceinline__ v8f vzero8() {
    v8f z = {0.f,0.f,0.f,0.f,0.f,0.f,0.f,0.f};
    return z;
}

// A-fragment (16x32 bf16, row-major source, leading dim = ld elements):
// lane m = lane&15, K-group g = lane>>4: elements = K g*16..g*16+15 (contiguous).
__device__ __forceinline__ v16bf load_a_frag(const bf16_t* base, size_t ld, int lane) {
    int m  = lane & 15;
    int kb = (lane >> 4) << 4;
    const bf16_t* p = base + (size_t)m * ld + kb;
    union { v8bf h[2]; v16bf v; } u;
    u.h[0] = *(const v8bf*)(p);        // 16B contiguous
    u.h[1] = *(const v8bf*)(p + 8);    // 16B contiguous
    return u.v;
}

// B-fragment (32x16 bf16) from K-major storage (Bt[n][k], leading dim = ld):
// lane n = lane&15, K-group g = lane>>4: VGPR v = K g*16 + {2v,2v+1}.
// One contiguous 32B load per lane.
__device__ __forceinline__ v16bf load_b_frag(const bf16_t* base, size_t ld, int lane) {
    int n  = lane & 15;
    int kb = (lane >> 4) << 4;
    return *(const v16bf*)(base + (size_t)n * ld + kb);
}

#define WMMA_BF16(a, b, c) \
    __builtin_amdgcn_wmma_f32_16x16x32_bf16(false, (a), false, (b), (short)0, (c), false, false)

// ---------------- utility kernels ----------------
__global__ void k_zero_f32(float* __restrict__ p, long count) {
    long i = (long)blockIdx.x * blockDim.x + threadIdx.x;
    if (i < count) p[i] = 0.0f;
}

__global__ void k_f32_to_bf16(const float* __restrict__ in, bf16_t* __restrict__ out, long count) {
    long i = (long)blockIdx.x * blockDim.x + threadIdx.x;
    if (i < count) out[i] = (bf16_t)in[i];
}

// transpose 512x512 fp32 -> bf16 [N][K] (K-major for B-fragments)
__global__ void k_transpose_w(const float* __restrict__ Wq, const float* __restrict__ Wk,
                              const float* __restrict__ Wv, const float* __restrict__ Wp,
                              bf16_t* __restrict__ tq, bf16_t* __restrict__ tk,
                              bf16_t* __restrict__ tv, bf16_t* __restrict__ tp) {
    const float* src; bf16_t* dst;
    switch (blockIdx.z) {
        case 0:  src = Wq; dst = tq; break;
        case 1:  src = Wk; dst = tk; break;
        case 2:  src = Wv; dst = tv; break;
        default: src = Wp; dst = tp; break;
    }
    __shared__ float tile[32][33];
    int x0 = blockIdx.x * 32, y0 = blockIdx.y * 32;
    int tx = threadIdx.x & 31, ty = threadIdx.x >> 5;   // 32 x 8
    #pragma unroll
    for (int i = 0; i < 32; i += 8)
        tile[ty + i][tx] = src[(size_t)(y0 + ty + i) * FF + x0 + tx];
    __syncthreads();
    #pragma unroll
    for (int i = 0; i < 32; i += 8)
        dst[(size_t)(x0 + ty + i) * FF + y0 + tx] = (bf16_t)tile[tx][ty + i];
}

// ---------------- generic 512-wide GEMM: C[M,512] = A[M,512] * Bt^T (+bias) ----------------
// grid: (M/128, 512/64); block 256 threads = 8 waves; wave -> 32x32 patch (2x2 tiles)
__global__ void k_gemm512(const bf16_t* __restrict__ A, const bf16_t* __restrict__ Bt,
                          float* __restrict__ C, const float* __restrict__ bias) {
    int row0 = blockIdx.x * 128;
    int col0 = blockIdx.y * 64;
    int w = threadIdx.x >> 5, lane = threadIdx.x & 31;
    int wm = (w & 3) * 32;
    int wn = (w >> 2) * 32;
    const bf16_t* Abase = A + (size_t)(row0 + wm) * FF;
    const bf16_t* Bbase = Bt + (size_t)(col0 + wn) * FF;

    v8f acc00 = vzero8(), acc01 = vzero8(), acc10 = vzero8(), acc11 = vzero8();
    for (int k0 = 0; k0 < FF; k0 += 32) {
        __builtin_prefetch((const void*)(Abase + k0 + 256), 0, 3);   // WGP-scope
        v16bf a0 = load_a_frag(Abase + k0, FF, lane);
        v16bf a1 = load_a_frag(Abase + (size_t)16 * FF + k0, FF, lane);
        v16bf b0 = load_b_frag(Bbase + k0, FF, lane);
        v16bf b1 = load_b_frag(Bbase + (size_t)16 * FF + k0, FF, lane);
        acc00 = WMMA_BF16(a0, b0, acc00);
        acc01 = WMMA_BF16(a0, b1, acc01);
        acc10 = WMMA_BF16(a1, b0, acc10);
        acc11 = WMMA_BF16(a1, b1, acc11);
    }
    int nl = lane & 15, mo = (lane >> 4) * 8;
    #pragma unroll
    for (int i = 0; i < 2; i++) {
        #pragma unroll
        for (int j = 0; j < 2; j++) {
            const v8f& acc = (i == 0) ? (j == 0 ? acc00 : acc01) : (j == 0 ? acc10 : acc11);
            int col = col0 + wn + j * 16 + nl;
            float bv = bias ? bias[col] : 0.0f;
            float* cptr = C + (size_t)(row0 + wm + i * 16 + mo) * FF + col;
            #pragma unroll
            for (int r = 0; r < 8; r++)
                cptr[(size_t)r * FF] = acc[r] + bv;
        }
    }
}

// ---------------- column sum-of-squares of q_inp/k_inp over n (split + atomic) ----------------
// grid (FF/64, BB*NCHUNK, 2), block 256 = 64 cols x 4 n-slices; accumulates into qn/kn
__global__ void k_colnorms(const float* __restrict__ q, const float* __restrict__ k,
                           float* __restrict__ qn, float* __restrict__ kn) {
    int tid = threadIdx.x;
    int col = blockIdx.x * 64 + (tid & 63);
    int sub = tid >> 6;
    int b = blockIdx.y / NCHUNK;
    int chunk = blockIdx.y % NCHUNK;
    int n0 = chunk * (NN / NCHUNK);
    const float* src = blockIdx.z ? k : q;
    const float* p = src + (size_t)(b * NN + n0) * FF + col;
    float s = 0.f;
    for (int n = sub; n < NN / NCHUNK; n += 4) { float v = p[(size_t)n * FF]; s += v * v; }
    __shared__ float red[256];
    red[tid] = s;
    __syncthreads();
    if (sub == 0) {
        float t = red[tid] + red[tid + 64] + red[tid + 128] + red[tid + 192];
        float* dst = blockIdx.z ? kn : qn;
        atomicAdd(&dst[b * FF + col], t);
    }
}

// ---------------- build normalized qT/kT bf16 in [ (b*FF + f), n ] layout ----------------
// LDS tile transpose: coalesced read along f, coalesced write along n.
// row index (b*4+h)*128+dh == b*512 + f.  grid (NN/32, FF/32, BB*2), block 256
__global__ void k_build_qkT(const float* __restrict__ q_inp, const float* __restrict__ k_inp,
                            const float* __restrict__ qn, const float* __restrict__ kn,
                            bf16_t* __restrict__ qT, bf16_t* __restrict__ kT) {
    int b = blockIdx.z & 1;
    int which = blockIdx.z >> 1;
    const float* src = which ? k_inp : q_inp;
    const float* ns  = which ? kn : qn;
    bf16_t* dst = which ? kT : qT;

    __shared__ float tile[32][33];
    int n0 = blockIdx.x * 32, f0 = blockIdx.y * 32;
    int tx = threadIdx.x & 31, ty = threadIdx.x >> 5;   // 32 x 8
    #pragma unroll
    for (int i = 0; i < 32; i += 8)
        tile[ty + i][tx] = src[(size_t)(b * NN + n0 + ty + i) * FF + f0 + tx];
    __syncthreads();
    #pragma unroll
    for (int i = 0; i < 32; i += 8) {
        int f = f0 + ty + i;
        float inv = 1.0f / fmaxf(sqrtf(ns[b * FF + f]), 1e-12f);
        dst[(size_t)(b * FF + f) * NN + n0 + tx] = (bf16_t)(tile[tx][ty + i] * inv);
    }
}

// ---------------- attn partials: split-K (k_norm @ q_norm^T), atomic accumulation ----------
// grid (NN/KSPAN, 1, BB*HEADS) = (16,1,8); block 256 = 8 waves; wave -> 32 rows x 64 cols
__global__ void k_attn_gemm(const bf16_t* __restrict__ kT, const bf16_t* __restrict__ qT,
                            float* __restrict__ attn) {
    int bh = blockIdx.z;
    int kbase = blockIdx.x * KSPAN;
    const bf16_t* A  = kT + (size_t)bh * DH * NN + kbase;   // [d, n]
    const bf16_t* Bm = qT + (size_t)bh * DH * NN + kbase;   // [e, n] -> column-major B
    int w = threadIdx.x >> 5, lane = threadIdx.x & 31;
    int m0 = (w >> 1) * 32;
    int e0 = (w & 1) * 64;

    v8f acc[2][4];
    #pragma unroll
    for (int i = 0; i < 2; i++)
        #pragma unroll
        for (int j = 0; j < 4; j++) acc[i][j] = vzero8();

    for (int kk = 0; kk < KSPAN; kk += 32) {
        __builtin_prefetch((const void*)(A + (size_t)m0 * NN + kk + 512), 0, 3);
        v16bf a0 = load_a_frag(A + (size_t)(m0)      * NN + kk, NN, lane);
        v16bf a1 = load_a_frag(A + (size_t)(m0 + 16) * NN + kk, NN, lane);
        #pragma unroll
        for (int j = 0; j < 4; j++) {
            v16bf bf = load_b_frag(Bm + (size_t)(e0 + j * 16) * NN + kk, NN, lane);
            acc[0][j] = WMMA_BF16(a0, bf, acc[0][j]);
            acc[1][j] = WMMA_BF16(a1, bf, acc[1][j]);
        }
    }
    int nl = lane & 15, mo = (lane >> 4) * 8;
    #pragma unroll
    for (int i = 0; i < 2; i++)
        #pragma unroll
        for (int j = 0; j < 4; j++) {
            int dbase = m0 + i * 16 + mo;
            int e = e0 + j * 16 + nl;
            float* ap = attn + (size_t)bh * DH * DH + (size_t)dbase * DH + e;
            #pragma unroll
            for (int r = 0; r < 8; r++)
                atomicAdd(ap + (size_t)r * DH, acc[i][j][r]);   // no-return f32 atomic
        }
}

// ---------------- row softmax over 128 (applies rescale), fp32 -> bf16 ----------------
__global__ void k_softmax(const float* __restrict__ attn, const float* __restrict__ rescale,
                          bf16_t* __restrict__ attn_b) {
    int row = blockIdx.x;           // BB*HEADS*128 rows
    int h = (row >> 7) & (HEADS - 1);
    int tid = threadIdx.x;          // 128 threads
    const float* base = attn + (size_t)row * DH;
    __shared__ float red[128];
    float v = base[tid] * rescale[h];
    red[tid] = v;
    __syncthreads();
    for (int s = 64; s > 0; s >>= 1) { if (tid < s) red[tid] = fmaxf(red[tid], red[tid + s]); __syncthreads(); }
    float mx = red[0];
    __syncthreads();
    float e = __expf(v - mx);
    red[tid] = e;
    __syncthreads();
    for (int s = 64; s > 0; s >>= 1) { if (tid < s) red[tid] += red[tid + s]; __syncthreads(); }
    attn_b[(size_t)row * DH + tid] = (bf16_t)(e / red[0]);
}

// ---------------- out = attn_b @ v   -> oatt bf16 [b, n, h*128+d] ----------------
// per (b,h): M=128 (d), Ncols=16384 (spatial), K=128 (e)
// grid (NN/64, 1, 8); block 256 = 8 waves; wave -> 32 rows x 32 cols (2x2 tiles)
__global__ void k_out_gemm(const bf16_t* __restrict__ attn_b, const bf16_t* __restrict__ v_bf,
                           bf16_t* __restrict__ oatt) {
    int bh = blockIdx.z;
    int b = bh >> 2, h = bh & 3;
    int col0 = blockIdx.x * 64;
    int w = threadIdx.x >> 5, lane = threadIdx.x & 31;
    int wm = (w & 3) * 32;          // d rows
    int wn = (w >> 2) * 32;         // spatial cols
    const bf16_t* A = attn_b + (size_t)bh * DH * DH;
    const bf16_t* Vb = v_bf + (size_t)(b * NN + col0 + wn) * FF + h * DH;

    v8f acc[2][2];
    #pragma unroll
    for (int i = 0; i < 2; i++)
        #pragma unroll
        for (int j = 0; j < 2; j++) acc[i][j] = vzero8();

    #pragma unroll
    for (int kk = 0; kk < DH; kk += 32) {
        v16bf a0 = load_a_frag(A + (size_t)(wm)      * DH + kk, DH, lane);
        v16bf a1 = load_a_frag(A + (size_t)(wm + 16) * DH + kk, DH, lane);
        v16bf b0 = load_b_frag(Vb + kk, FF, lane);
        v16bf b1 = load_b_frag(Vb + (size_t)16 * FF + kk, FF, lane);
        acc[0][0] = WMMA_BF16(a0, b0, acc[0][0]);
        acc[0][1] = WMMA_BF16(a0, b1, acc[0][1]);
        acc[1][0] = WMMA_BF16(a1, b0, acc[1][0]);
        acc[1][1] = WMMA_BF16(a1, b1, acc[1][1]);
    }
    int nl = lane & 15, mo = (lane >> 4) * 8;
    #pragma unroll
    for (int i = 0; i < 2; i++)
        #pragma unroll
        for (int j = 0; j < 2; j++) {
            int dbase = wm + i * 16 + mo;
            int ncol = col0 + wn + j * 16 + nl;
            bf16_t* op = oatt + (size_t)(b * NN + ncol) * FF + h * DH + dbase;
            #pragma unroll
            for (int r = 0; r < 8; r++)
                op[r] = (bf16_t)acc[i][j][r];
        }
}

// ---------------- depthwise 3x3x3 conv + GeLU (positional branch, fp32) ----------------
// vp[b,c,z,y,x] = v_inp[(b*NN + y*W + x)*FF + z*64 + c];  p1 laid out [b][c][z][y][x]
__global__ void k_dwconv1(const float* __restrict__ v_inp, const float* __restrict__ w1,
                          float* __restrict__ p1) {
    int idx = blockIdx.x * 256 + threadIdx.x;
    int x = idx & 127;
    int y = (idx >> 7) & 127;
    int z = (idx >> 14) & 7;
    int c = (idx >> 17) & 63;
    int b = idx >> 23;
    float acc = 0.f;
    #pragma unroll
    for (int kd = -1; kd <= 1; kd++) {
        int zz = z + kd; if ((unsigned)zz >= (unsigned)DD) continue;
        #pragma unroll
        for (int kh = -1; kh <= 1; kh++) {
            int yy = y + kh; if ((unsigned)yy >= (unsigned)HH) continue;
            #pragma unroll
            for (int kw = -1; kw <= 1; kw++) {
                int xx = x + kw; if ((unsigned)xx >= (unsigned)WW) continue;
                float v = v_inp[(size_t)(b * NN + yy * WW + xx) * FF + zz * CC + c];
                acc += v * w1[c * 27 + (kd + 1) * 9 + (kh + 1) * 3 + (kw + 1)];
            }
        }
    }
    p1[idx] = 0.5f * acc * (1.0f + erff(acc * 0.70710678118654752f));
}

__global__ void k_dwconv2_add(const float* __restrict__ p1, const float* __restrict__ w2,
                              float* __restrict__ outp) {
    int idx = blockIdx.x * 256 + threadIdx.x;
    int x = idx & 127;
    int y = (idx >> 7) & 127;
    int z = (idx >> 14) & 7;
    int c = (idx >> 17) & 63;
    int b = idx >> 23;
    float acc = 0.f;
    #pragma unroll
    for (int kd = -1; kd <= 1; kd++) {
        int zz = z + kd; if ((unsigned)zz >= (unsigned)DD) continue;
        #pragma unroll
        for (int kh = -1; kh <= 1; kh++) {
            int yy = y + kh; if ((unsigned)yy >= (unsigned)HH) continue;
            #pragma unroll
            for (int kw = -1; kw <= 1; kw++) {
                int xx = x + kw; if ((unsigned)xx >= (unsigned)WW) continue;
                float v = p1[((size_t)((b * CC + c) * DD + zz) << 14) + yy * WW + xx];
                acc += v * w2[c * 27 + (kd + 1) * 9 + (kh + 1) * 3 + (kw + 1)];
            }
        }
    }
    outp[(size_t)(b * NN + y * WW + x) * FF + z * CC + c] += acc;
}

// ---------------- host-side launch ----------------
extern "C" void kernel_launch(void* const* d_in, const int* in_sizes, int n_in,
                              void* d_out, int out_size, void* d_ws, size_t ws_size,
                              hipStream_t stream) {
    const float* x_in    = (const float*)d_in[0];
    const float* Wq      = (const float*)d_in[1];
    const float* Wk      = (const float*)d_in[2];
    const float* Wv      = (const float*)d_in[3];
    const float* rescale = (const float*)d_in[4];
    const float* Wp      = (const float*)d_in[5];
    const float* bp      = (const float*)d_in[6];
    const float* pos_w1  = (const float*)d_in[7];
    const float* pos_w2  = (const float*)d_in[8];
    float* out = (float*)d_out;

    // workspace carve-out (256B aligned)
    char* base = (char*)d_ws;
    size_t off = 0;
    auto take = [&](size_t bytes) -> void* {
        void* r = base + off;
        off = (off + bytes + 255) & ~(size_t)255;
        return r;
    };
    bf16_t* xb    = (bf16_t*)take(BNF * 2);
    bf16_t* wq_t  = (bf16_t*)take((size_t)FF * FF * 2);
    bf16_t* wk_t  = (bf16_t*)take((size_t)FF * FF * 2);
    bf16_t* wv_t  = (bf16_t*)take((size_t)FF * FF * 2);
    bf16_t* wp_t  = (bf16_t*)take((size_t)FF * FF * 2);
    float*  q_inp = (float*)take(BNF * 4);
    float*  k_inp = (float*)take(BNF * 4);
    float*  v_inp = (float*)take(BNF * 4);
    float*  qn    = (float*)take((size_t)BB * FF * 4);
    float*  kn    = (float*)take((size_t)BB * FF * 4);
    bf16_t* qT    = (bf16_t*)take(BNF * 2);
    bf16_t* kT    = (bf16_t*)take(BNF * 2);
    bf16_t* v_bf  = (bf16_t*)take(BNF * 2);
    float*  attn  = (float*)take((size_t)BB * HEADS * DH * DH * 4);
    bf16_t* attn_b= (bf16_t*)take((size_t)BB * HEADS * DH * DH * 2);
    // aliases (lifetimes do not overlap):
    float*  p1    = q_inp;            // written after q_inp last read (k_build_qkT)
    bf16_t* oatt  = (bf16_t*)k_inp;   // written after k_inp last read (k_build_qkT)

    const int CONV_BLKS = (int)(BNF / 256);   // 65536
    const long ATTN_ELEMS = (long)BB * HEADS * DH * DH;   // 131072

    // 0) zero the atomic accumulators (norms + attn)
    k_zero_f32<<<(int)((2 * BB * FF + 255) / 256), 256, 0, stream>>>(qn, 2L * BB * FF); // qn+kn contiguous
    k_zero_f32<<<(int)((ATTN_ELEMS + 255) / 256), 256, 0, stream>>>(attn, ATTN_ELEMS);
    // 1) x -> bf16
    k_f32_to_bf16<<<CONV_BLKS, 256, 0, stream>>>(x_in, xb, (long)BNF);
    // 2) weights -> bf16, K-major
    k_transpose_w<<<dim3(16, 16, 4), 256, 0, stream>>>(Wq, Wk, Wv, Wp, wq_t, wk_t, wv_t, wp_t);
    // 3) QKV GEMMs
    dim3 ggrid(BN / 128, FF / 64, 1);
    k_gemm512<<<ggrid, 256, 0, stream>>>(xb, wq_t, q_inp, nullptr);
    k_gemm512<<<ggrid, 256, 0, stream>>>(xb, wk_t, k_inp, nullptr);
    k_gemm512<<<ggrid, 256, 0, stream>>>(xb, wv_t, v_inp, nullptr);
    // 4) column sum-of-squares over n (split across NCHUNK, atomic accumulate)
    k_colnorms<<<dim3(FF / 64, BB * NCHUNK, 2), 256, 0, stream>>>(q_inp, k_inp, qn, kn);
    // 5) normalized qT/kT bf16 [b*FF + f][n]  (LDS tile transpose, coalesced both sides)
    k_build_qkT<<<dim3(NN / 32, FF / 32, BB * 2), 256, 0, stream>>>(q_inp, k_inp, qn, kn, qT, kT);
    // 6) v -> bf16 (kept in natural [n, f] layout, serves as B for attn@v)
    k_f32_to_bf16<<<CONV_BLKS, 256, 0, stream>>>(v_inp, v_bf, (long)BNF);
    // 7) channel attention logits, split-K with f32 atomics
    k_attn_gemm<<<dim3(NN / KSPAN, 1, BB * HEADS), 256, 0, stream>>>(kT, qT, attn);
    // 8) softmax (applies rescale)
    k_softmax<<<BB * HEADS * DH, 128, 0, stream>>>(attn, rescale, attn_b);
    // 9) attn @ v -> oatt bf16 [b, n, h*128+d]
    k_out_gemm<<<dim3(NN / 64, 1, BB * HEADS), 256, 0, stream>>>(attn_b, v_bf, oatt);
    // 10) projection + bias -> d_out (fp32)
    k_gemm512<<<ggrid, 256, 0, stream>>>(oatt, wp_t, out, bp);
    // 11) positional branch: dwconv -> gelu -> dwconv, accumulate into d_out
    k_dwconv1<<<CONV_BLKS, 256, 0, stream>>>(v_inp, pos_w1, p1);
    k_dwconv2_add<<<CONV_BLKS, 256, 0, stream>>>(p1, pos_w2, out);
}
